// PerformerAttention_71038759076381
// MI455X (gfx1250) — matrix operand order repaired and checked by reading, not compiled, not run on hardware
//
#include <hip/hip_runtime.h>
#include <hip/hip_bf16.h>

typedef __attribute__((ext_vector_type(16))) _Float16 v16h;
typedef __attribute__((ext_vector_type(8)))  float    v8f;

#define B_   4
#define H_   16
#define S_   4096
#define D_   64
#define M_   256
#define BH_  64
#define SCALE 0.125f     // D^-0.25 (outer) * D^-0.25 (inner data_normalizer) = D^-0.5
#define RATIO 0.0625f    // M^-0.5
#define EPSV  1.0e-4f
#define KITER 8

union FragA { v16h v; unsigned u[8]; _Float16 h[16]; };
union FragC { v8f  v; float f[8]; };

__device__ __forceinline__ unsigned pack2(float a, float b) {
  union { unsigned u; _Float16 h[2]; } p;
  p.h[0] = (_Float16)a; p.h[1] = (_Float16)b;
  return p.u;
}

__device__ __forceinline__ void wmma_f16(FragC& c, const FragA& a, const FragA& b) {
  c.v = __builtin_amdgcn_wmma_f32_16x16x32_f16(false, a.v, false, b.v, (short)0, c.v, false, false);
}

// A-matrix 16x32 f16 fragment from row-major [row][k] (stride in halves; pairs 4B aligned).
// Lane l: M = row0 + (l&15); half h = l>>4; v0..3 hold K = kb+8h+{0..7}; v4..7 hold K = kb+8h+16+{0..7}.
__device__ __forceinline__ void load_a(const _Float16* base, int stride, int row0, int kb,
                                       int lane, FragA& A) {
  const _Float16* r = base + (row0 + (lane & 15)) * stride + kb + 8 * (lane >> 4);
  #pragma unroll
  for (int i = 0; i < 4; i++) A.u[i]     = *(const unsigned*)(r + 2 * i);
  #pragma unroll
  for (int i = 0; i < 4; i++) A.u[4 + i] = *(const unsigned*)(r + 16 + 2 * i);
}

// B-matrix 32x16 f16 fragment; element B[k][n] stored at base[n*stride + k] (k contiguous).
// Lane l: N = n0 + (l&15); half h = l>>4; v0..3 hold K = kb+8h+{0..7}; v4..7 hold K = kb+8h+16+{0..7}.
__device__ __forceinline__ void load_b(const _Float16* base, int stride, int n0, int kb,
                                       int lane, FragA& Bf) {
  const _Float16* r = base + (n0 + (lane & 15)) * stride + kb + 8 * (lane >> 4);
  #pragma unroll
  for (int i = 0; i < 4; i++) Bf.u[i]     = *(const unsigned*)(r + 2 * i);
  #pragma unroll
  for (int i = 0; i < 4; i++) Bf.u[4 + i] = *(const unsigned*)(r + 16 + 2 * i);
}

__device__ __forceinline__ void atomicMaxF(float* addr, float v) {
  unsigned* ua = (unsigned*)addr;
  unsigned cur = __float_as_uint(*addr);
  while (__uint_as_float(cur) < v) {
    unsigned assumed = cur;
    cur = atomicCAS(ua, assumed, __float_as_uint(v));
    if (cur == assumed) break;
  }
}

// ---------------- kernel 0: init workspace, convert projection to f16 ----------------
__global__ void k_init(const float* __restrict__ proj, float* gmax, float* kcum,
                       float* ctx, _Float16* projh) {
  int idx = blockIdx.x * blockDim.x + threadIdx.x;
  if (idx < BH_ * M_ * D_) ctx[idx] = 0.f;
  if (idx < BH_ * M_)      kcum[idx] = 0.f;
  if (idx < (M_ * D_) / 2) {
    *(unsigned*)&projh[2 * idx] = pack2(proj[2 * idx], proj[2 * idx + 1]);
  }
  if (idx == 0) gmax[0] = -3.0e38f;
}

// ---------------- kernel 1: global max of key-side data_dash ----------------
__global__ __launch_bounds__(128) void k_kmax(const float* __restrict__ Kg,
                                              const float* __restrict__ maskg,
                                              const _Float16* __restrict__ projh,
                                              float* gmax) {
  __shared__ _Float16 sK[64][66];
  __shared__ float smask[64];
  __shared__ float red[128];
  int tid = threadIdx.x, lane = tid & 31, wid = tid >> 5;
  int bh = blockIdx.x, b = bh >> 4;
  float gm = -3.0e38f;
  for (int it = 0; it < KITER; ++it) {
    int s0 = (blockIdx.y * KITER + it) * 64;
    if (tid < 64) smask[tid] = maskg[b * S_ + s0 + tid];
    if (it + 1 < KITER && tid < 64) {       // prefetch next S-block of K
      const float* nxt = &Kg[(bh * S_ + s0 + 64 + tid) * D_];
      __builtin_prefetch(nxt, 0, 3);
      __builtin_prefetch(nxt + 32, 0, 3);
    }
    __syncthreads();
    for (int idx = tid; idx < 64 * 16; idx += 128) {   // float4 loads, packed f16 stores
      int r = idx >> 4, c4 = (idx & 15) * 4;
      float4 kv = *(const float4*)&Kg[(bh * S_ + s0 + r) * D_ + c4];
      float sc = SCALE * smask[r];
      *(unsigned*)&sK[r][c4]     = pack2(kv.x * sc, kv.y * sc);
      *(unsigned*)&sK[r][c4 + 2] = pack2(kv.z * sc, kv.w * sc);
    }
    __syncthreads();
    FragA a0, a1;
    load_a(&sK[0][0], 66, wid * 16, 0,  lane, a0);
    load_a(&sK[0][0], 66, wid * 16, 32, lane, a1);
    #pragma unroll
    for (int t = 0; t < 16; t++) {
      FragA b0, b1; FragC acc;
      #pragma unroll
      for (int j = 0; j < 8; j++) acc.f[j] = 0.f;
      load_b(projh, 64, t * 16, 0,  lane, b0);
      load_b(projh, 64, t * 16, 32, lane, b1);
      wmma_f16(acc, a0, b0);
      wmma_f16(acc, a1, b1);
      #pragma unroll
      for (int j = 0; j < 8; j++) gm = fmaxf(gm, acc.f[j]);
    }
    __syncthreads();
  }
  red[tid] = gm;
  __syncthreads();
  for (int off = 64; off; off >>= 1) {
    if (tid < off) red[tid] = fmaxf(red[tid], red[tid + off]);
    __syncthreads();
  }
  if (tid == 0) atomicMaxF(gmax, red[0]);
}

// ---------------- kernel 2: kp = exp(...), k_cumsum, context = kp^T @ v ----------------
__global__ __launch_bounds__(128) void k_kp(const float* __restrict__ Kg,
                                            const float* __restrict__ Vg,
                                            const float* __restrict__ maskg,
                                            const _Float16* __restrict__ projh,
                                            const float* __restrict__ gmaxp,
                                            float* kcum, float* ctx) {
  __shared__ _Float16 sK[64][66];
  __shared__ _Float16 sVT[64][66];   // [d][s] so B pairs along K=s are contiguous
  __shared__ _Float16 kpT[256][66];  // [m][s] for A-fragment loads of GEMM2
  __shared__ float smask[64];
  __shared__ float sdiag[64];
  int tid = threadIdx.x, lane = tid & 31, wid = tid >> 5;
  int bh = blockIdx.x, b = bh >> 4;
  float gmax = gmaxp[0];
  FragC cacc[4][4];
  #pragma unroll
  for (int r = 0; r < 4; r++)
    #pragma unroll
    for (int c = 0; c < 4; c++)
      #pragma unroll
      for (int j = 0; j < 8; j++) cacc[r][c].f[j] = 0.f;
  float kc[16];
  #pragma unroll
  for (int t = 0; t < 16; t++) kc[t] = 0.f;

  for (int it = 0; it < KITER; ++it) {
    int s0 = (blockIdx.y * KITER + it) * 64;
    if (tid < 64) smask[tid] = maskg[b * S_ + s0 + tid];
    if (it + 1 < KITER) {                 // prefetch next S-block of K and V
      int r = tid & 63;
      const float* nxt = (tid < 64) ? &Kg[(bh * S_ + s0 + 64 + r) * D_]
                                    : &Vg[(bh * S_ + s0 + 64 + r) * D_];
      __builtin_prefetch(nxt, 0, 3);
      __builtin_prefetch(nxt + 32, 0, 3);
    }
    __syncthreads();
    for (int idx = tid; idx < 64 * 16; idx += 128) {
      int r = idx >> 4, c4 = (idx & 15) * 4;
      float mv = smask[r];
      float4 kv = *(const float4*)&Kg[(bh * S_ + s0 + r) * D_ + c4];
      float sc = SCALE * mv;
      *(unsigned*)&sK[r][c4]     = pack2(kv.x * sc, kv.y * sc);
      *(unsigned*)&sK[r][c4 + 2] = pack2(kv.z * sc, kv.w * sc);
      float4 vv = *(const float4*)&Vg[(bh * S_ + s0 + r) * D_ + c4];
      sVT[c4][r]     = (_Float16)(vv.x * mv);
      sVT[c4 + 1][r] = (_Float16)(vv.y * mv);
      sVT[c4 + 2][r] = (_Float16)(vv.z * mv);
      sVT[c4 + 3][r] = (_Float16)(vv.w * mv);
    }
    __syncthreads();
    if (tid < 64) {
      float s = 0.f;
      for (int c = 0; c < 64; c++) { float x = (float)sK[tid][c]; s += x * x; }
      sdiag[tid] = 0.5f * s;
    }
    __syncthreads();

    // GEMM1: dd strip (16 s-rows x 256 m) per wave; exp -> kpT; kcum partials.
    int strip = wid * 16;
    FragA a0, a1;
    load_a(&sK[0][0], 66, strip, 0,  lane, a0);
    load_a(&sK[0][0], 66, strip, 32, lane, a1);
    float dg[8];
    #pragma unroll
    for (int j = 0; j < 8; j++) dg[j] = sdiag[strip + j + 8 * (lane >> 4)];
    #pragma unroll
    for (int t = 0; t < 16; t++) {
      FragA b0, b1; FragC acc;
      #pragma unroll
      for (int j = 0; j < 8; j++) acc.f[j] = 0.f;
      load_b(projh, 64, t * 16, 0,  lane, b0);
      load_b(projh, 64, t * 16, 32, lane, b1);
      wmma_f16(acc, a0, b0);
      wmma_f16(acc, a1, b1);
      float colsum = 0.f;
      #pragma unroll
      for (int j = 0; j < 8; j++) {
        float v = RATIO * (__expf(acc.f[j] - dg[j] - gmax) + EPSV);
        colsum += v;
        kpT[t * 16 + (lane & 15)][strip + j + 8 * (lane >> 4)] = (_Float16)v;
      }
      colsum += __shfl_xor(colsum, 16, 32);  // combine both 8-row halves (same m column)
      kc[t] += colsum;
    }
    __syncthreads();

    // GEMM2: context strip (64 m-rows x 64 d-cols) per wave, contraction over 64 s-rows.
    int m0 = wid * 64;
    #pragma unroll
    for (int kk = 0; kk < 2; kk++) {
      FragA av[4], bv[4];
      #pragma unroll
      for (int r = 0; r < 4; r++) load_a(&kpT[0][0], 66, m0 + 16 * r, kk * 32, lane, av[r]);
      #pragma unroll
      for (int c = 0; c < 4; c++) load_b(&sVT[0][0], 66, c * 16, kk * 32, lane, bv[c]);
      #pragma unroll
      for (int r = 0; r < 4; r++)
        #pragma unroll
        for (int c = 0; c < 4; c++) wmma_f16(cacc[r][c], av[r], bv[c]);
    }
    __syncthreads();
  }

  // Flush per-WG partials.
  int m0 = wid * 64;
  #pragma unroll
  for (int r = 0; r < 4; r++)
    #pragma unroll
    for (int c = 0; c < 4; c++)
      #pragma unroll
      for (int j = 0; j < 8; j++) {
        int m = m0 + 16 * r + j + 8 * (lane >> 4);
        int d = 16 * c + (lane & 15);
        atomicAdd(&ctx[bh * M_ * D_ + m * D_ + d], cacc[r][c].f[j]);
      }
  if (lane < 16) {
    #pragma unroll
    for (int t = 0; t < 16; t++)
      atomicAdd(&kcum[bh * M_ + t * 16 + lane], kc[t]);
  }
}

// ---------------- kernel 3: context -> transposed f16 [bh][d][m] (packed stores) ----------------
__global__ void k_ctxT(const float* __restrict__ ctx, _Float16* ctxTh) {
  int p = blockIdx.x * blockDim.x + threadIdx.x;       // one f16 pair per thread
  if (p >= BH_ * D_ * (M_ / 2)) return;
  int bh = p / (D_ * (M_ / 2));
  int rem = p - bh * (D_ * (M_ / 2));
  int d = rem / (M_ / 2);
  int m = (rem - d * (M_ / 2)) * 2;
  float a = ctx[bh * M_ * D_ + m * D_ + d];
  float b2 = ctx[bh * M_ * D_ + (m + 1) * D_ + d];
  *(unsigned*)&ctxTh[bh * (D_ * M_) + d * M_ + m] = pack2(a, b2);
}

// ---------------- kernel 4: qp, D_inv, out = D_inv * (qp @ context) ----------------
__global__ __launch_bounds__(128) void k_qout(const float* __restrict__ Qg,
                                              const _Float16* __restrict__ projh,
                                              const float* __restrict__ kcum,
                                              const _Float16* __restrict__ ctxTh,
                                              float* __restrict__ out) {
  __shared__ _Float16 sQ[64][66];
  __shared__ _Float16 qp[64][258];   // [s][m], even stride for 4B pair loads
  __shared__ float sdiag[64];
  __shared__ float sdinv[64];
  __shared__ float skc[256];
  int tid = threadIdx.x, lane = tid & 31, wid = tid >> 5;
  int bh = blockIdx.x;
  int s0 = blockIdx.y * 64;

  for (int idx = tid; idx < 64 * 16; idx += 128) {
    int r = idx >> 4, c4 = (idx & 15) * 4;
    float4 qv = *(const float4*)&Qg[(bh * S_ + s0 + r) * D_ + c4];
    *(unsigned*)&sQ[r][c4]     = pack2(qv.x * SCALE, qv.y * SCALE);
    *(unsigned*)&sQ[r][c4 + 2] = pack2(qv.z * SCALE, qv.w * SCALE);
  }
  for (int i = tid; i < 256; i += 128) skc[i] = kcum[bh * M_ + i];
  __syncthreads();
  if (tid < 64) {
    float s = 0.f;
    for (int c = 0; c < 64; c++) { float x = (float)sQ[tid][c]; s += x * x; }
    sdiag[tid] = 0.5f * s;
  }
  __syncthreads();

  int strip = wid * 16;
  FragA a0, a1;
  load_a(&sQ[0][0], 66, strip, 0,  lane, a0);
  load_a(&sQ[0][0], 66, strip, 32, lane, a1);
  FragC acc[16];
  #pragma unroll
  for (int t = 0; t < 16; t++) {
    FragA b0, b1;
    #pragma unroll
    for (int j = 0; j < 8; j++) acc[t].f[j] = 0.f;
    load_b(projh, 64, t * 16, 0,  lane, b0);
    load_b(projh, 64, t * 16, 32, lane, b1);
    wmma_f16(acc[t], a0, b0);
    wmma_f16(acc[t], a1, b1);
  }
  // Per-row max: C-fragment row = j + 8*(lane>=16); reduce over tiles then 16-lane group.
  float rm[8];
  #pragma unroll
  for (int j = 0; j < 8; j++) {
    float m = -3.0e38f;
    #pragma unroll
    for (int t = 0; t < 16; t++) m = fmaxf(m, acc[t].f[j]);
    #pragma unroll
    for (int off = 1; off < 16; off <<= 1) m = fmaxf(m, __shfl_xor(m, off, 32));
    rm[j] = m;
  }
  float dg[8];
  #pragma unroll
  for (int j = 0; j < 8; j++) dg[j] = sdiag[strip + j + 8 * (lane >> 4)];
  #pragma unroll
  for (int t = 0; t < 16; t++)
    #pragma unroll
    for (int j = 0; j < 8; j++) {
      float v = RATIO * (__expf(acc[t].f[j] - dg[j] - rm[j]) + EPSV);
      qp[strip + j + 8 * (lane >> 4)][t * 16 + (lane & 15)] = (_Float16)v;
    }
  __syncthreads();
  {
    // D_inv: 2 threads per row, 128 elems each, combine via shfl_xor(1).
    int row = tid >> 1, half = tid & 1;
    float dot = 0.f;
    const _Float16* qr = &qp[row][half * 128];
    const float* kr = &skc[half * 128];
    for (int m = 0; m < 128; m++) dot += (float)qr[m] * kr[m];
    dot += __shfl_xor(dot, 1, 32);
    if (half == 0) sdinv[row] = 1.0f / dot;
  }
  __syncthreads();

  FragC oacc[4];
  #pragma unroll
  for (int c = 0; c < 4; c++)
    #pragma unroll
    for (int j = 0; j < 8; j++) oacc[c].f[j] = 0.f;
  const _Float16* ctxb = ctxTh + bh * (D_ * M_);
  #pragma unroll
  for (int kk = 0; kk < 8; kk++) {
    FragA aq;
    load_a(&qp[0][0], 258, strip, kk * 32, lane, aq);
    #pragma unroll
    for (int c = 0; c < 4; c++) {
      FragA bf;
      load_b(ctxb, 256, c * 16, kk * 32, lane, bf);
      wmma_f16(oacc[c], aq, bf);
    }
  }
  float dv[8];
  #pragma unroll
  for (int j = 0; j < 8; j++) dv[j] = sdinv[strip + j + 8 * (lane >> 4)];
  #pragma unroll
  for (int c = 0; c < 4; c++)
    #pragma unroll
    for (int j = 0; j < 8; j++) {
      int srow = strip + j + 8 * (lane >> 4);
      out[(bh * S_ + s0 + srow) * D_ + c * 16 + (lane & 15)] = oacc[c].f[j] * dv[j];
    }
}

extern "C" void kernel_launch(void* const* d_in, const int* in_sizes, int n_in,
                              void* d_out, int out_size, void* d_ws, size_t ws_size,
                              hipStream_t stream) {
  (void)in_sizes; (void)n_in; (void)out_size; (void)ws_size;
  const float* Q    = (const float*)d_in[0];
  const float* K    = (const float*)d_in[1];
  const float* V    = (const float*)d_in[2];
  const float* mask = (const float*)d_in[3];
  const float* proj = (const float*)d_in[4];
  float* out = (float*)d_out;

  // workspace layout (floats then halves): ~6.4 MB total
  float* wsf   = (float*)d_ws;
  float* gmax  = wsf;                     // 1 (padded to 16)
  float* kcum  = wsf + 16;                // BH*M
  float* ctx   = wsf + 16 + BH_ * M_;     // BH*M*D
  _Float16* projh = (_Float16*)(ctx + BH_ * M_ * D_); // M*D halves
  _Float16* ctxTh = projh + M_ * D_;                  // BH*D*M halves

  k_init<<<dim3((BH_ * M_ * D_ + 255) / 256), dim3(256), 0, stream>>>(proj, gmax, kcum, ctx, projh);
  k_kmax<<<dim3(BH_, S_ / (64 * KITER)), dim3(128), 0, stream>>>(K, mask, projh, gmax);
  k_kp  <<<dim3(BH_, S_ / (64 * KITER)), dim3(128), 0, stream>>>(K, V, mask, projh, gmax, kcum, ctx);
  k_ctxT<<<dim3((BH_ * D_ * (M_ / 2) + 255) / 256), dim3(256), 0, stream>>>(ctx, ctxTh);
  k_qout<<<dim3(BH_, S_ / 64), dim3(128), 0, stream>>>(Q, projh, kcum, ctxTh, out);
}
